// InteractionGNNBlock_7559142441635
// MI455X (gfx1250) — matrix operand order repaired and checked by reading, not compiled, not run on hardware
//
#include <hip/hip_runtime.h>

typedef __bf16 bf16;
typedef __attribute__((ext_vector_type(16))) __bf16 v16bf;
typedef __attribute__((ext_vector_type(8)))  __bf16 v8bf;
typedef __attribute__((ext_vector_type(8)))  float  v8f;

#define LATENT   128
#define HIDDEN   512
#define N_NODES  50000
#define N_EDGES  500000
#define LN_EPS   1e-5f

#define MT       4           // m-tiles of 16 rows per wave
#define BM       (MT * 16)   // 64 rows per block
#define NTHREADS 256         // 8 wave32 waves
#define XS       520         // bf16 activation stride (rows 16B-aligned, bank-spread)

// ---------------------------------------------------------------------------
// Scatter-add of edge features into destination nodes (segment_sum)
// ---------------------------------------------------------------------------
__global__ void zero_f32_kernel(float* __restrict__ p, int n) {
    int i = blockIdx.x * blockDim.x + threadIdx.x;
    if (i < n) p[i] = 0.0f;
}

__global__ void scatter_add_kernel(const float* __restrict__ edges,
                                   const int* __restrict__ dst,
                                   float* __restrict__ msgs) {
    int i = blockIdx.x * blockDim.x + threadIdx.x;   // over N_EDGES*LATENT
    if (i >= N_EDGES * LATENT) return;
    int e = i >> 7;            // /128
    int c = i & (LATENT - 1);
    atomicAdd(&msgs[(size_t)dst[e] * LATENT + c], edges[i]);
}

// ---------------------------------------------------------------------------
// Weight prep: fp32 W[din][dout] -> bf16 Wt[dout][din] (transposed, so each
// WMMA B-fragment is two 16B global reads per lane)
// ---------------------------------------------------------------------------
__global__ void wtr_kernel(const float* __restrict__ W, bf16* __restrict__ Wt,
                           int din, int dout) {
    int i = blockIdx.x * blockDim.x + threadIdx.x;
    if (i >= din * dout) return;
    int k = i / dout, n = i - k * dout;
    Wt[(size_t)n * din + k] = (bf16)W[i];
}

// ---------------------------------------------------------------------------
// One Linear layer for a 64-row tile, accumulators kept in registers.
//   acc[m][t] : m-tile m (rows 16m..16m+15), column tile (wave + 8t)
// Each B fragment (one L2 read) feeds MT=4 WMMAs.
// Fragment layouts per CDNA5 ISA 7.12.2 (wave32):
//  A 16x32 bf16 : lane L row L%16, K = {32*kt + 8*(L/16) + [0..7], +16..+23}
//  B 32x16 bf16 : lane L col L%16, K = {32*kt + 8*(L/16) + [0..7], +16..+23}
//  C/D 16x16 f32: elem r -> row r + 8*(L/16), col L%16
// Bias is folded into the accumulators at the end.
// ---------------------------------------------------------------------------
template<int NPER>
__device__ inline void gemm_bias(const bf16* __restrict__ Wt,
                                 const float* __restrict__ bias,
                                 const bf16* __restrict__ xbuf,
                                 int din, int tid, v8f (&acc)[MT][NPER]) {
    const int lane = tid & 31, wave = tid >> 5;
    const int half = lane >> 4, lm = lane & 15;

#pragma unroll
    for (int m = 0; m < MT; ++m)
#pragma unroll
        for (int t = 0; t < NPER; ++t)
#pragma unroll
            for (int r = 0; r < 8; ++r) acc[m][t][r] = 0.0f;

    const int nkt = din >> 5;
    for (int kt = 0; kt < nkt; ++kt) {
        const int kb = kt * 32 + half * 8;
        v16bf a[MT];
#pragma unroll
        for (int m = 0; m < MT; ++m) {
            const v8bf* pa0 = (const v8bf*)(xbuf + (m * 16 + lm) * XS + kb);
            const v8bf* pa1 = (const v8bf*)(xbuf + (m * 16 + lm) * XS + kb + 16);
#pragma unroll
            for (int i = 0; i < 8; ++i) { a[m][i] = (*pa0)[i]; a[m][i + 8] = (*pa1)[i]; }
        }
#pragma unroll
        for (int t = 0; t < NPER; ++t) {
            const int n = (wave + 8 * t) * 16 + lm;
            const bf16* wr = Wt + (size_t)n * din + kt * 32 + half * 8;
            const v8bf* pb0 = (const v8bf*)wr;
            const v8bf* pb1 = (const v8bf*)(wr + 16);
            v16bf b;
#pragma unroll
            for (int i = 0; i < 8; ++i) { b[i] = (*pb0)[i]; b[i + 8] = (*pb1)[i]; }
#pragma unroll
            for (int m = 0; m < MT; ++m)
                acc[m][t] = __builtin_amdgcn_wmma_f32_16x16x32_bf16(
                    false, a[m], false, b, (short)0, acc[m][t], false, false);
        }
    }

#pragma unroll
    for (int t = 0; t < NPER; ++t) {
        const float bv = bias[(wave + 8 * t) * 16 + lm];
#pragma unroll
        for (int m = 0; m < MT; ++m)
#pragma unroll
            for (int r = 0; r < 8; ++r) acc[m][t][r] += bv;
    }
}

// ---------------------------------------------------------------------------
// Register-resident LayerNorm + activation.
//  1) per-lane partials over this wave's columns, __shfl_xor reduce within
//     the 16-lane half-group (rows r+8*half live there)
//  2) 8-wave partials combined through tiny LDS, stats per row
//  3) normalize registers m-tile at a time (bounded reg pressure), then
//     bf16 -> xbuf (MODE 0) or fp32 + residual -> global (1 relu / 2 tanh)
// ---------------------------------------------------------------------------
template<int NPER, int MODE>
__device__ inline void ln_reg(const float* __restrict__ g,
                              const float* __restrict__ be,
                              int dout, v8f (&acc)[MT][NPER],
                              bf16* __restrict__ xbuf,
                              float (*part)[8], float (*part2)[8],
                              float (*stats)[2], int tid, int nvalid,
                              const float* __restrict__ resid,
                              float* __restrict__ outp) {
    const int lane = tid & 31, wave = tid >> 5;
    const int half = lane >> 4, lm = lane & 15;
    const float inv = 1.0f / (float)dout;

#pragma unroll
    for (int m = 0; m < MT; ++m)
#pragma unroll
        for (int r = 0; r < 8; ++r) {
            float s = 0.0f, s2 = 0.0f;
#pragma unroll
            for (int t = 0; t < NPER; ++t) {
                float v = acc[m][t][r];
                s += v; s2 += v * v;
            }
#pragma unroll
            for (int off = 8; off; off >>= 1) {
                s  += __shfl_xor(s,  off);
                s2 += __shfl_xor(s2, off);
            }
            if (lm == 0) {
                const int row = m * 16 + r + 8 * half;
                part[row][wave]  = s;
                part2[row][wave] = s2;
            }
        }
    __syncthreads();

    if (tid < BM) {
        float s = 0.0f, s2 = 0.0f;
#pragma unroll
        for (int w = 0; w < 8; ++w) { s += part[tid][w]; s2 += part2[tid][w]; }
        const float mean = s * inv;
        const float var  = s2 * inv - mean * mean;
        stats[tid][0] = mean;
        stats[tid][1] = rsqrtf(var + LN_EPS);
    }
    __syncthreads();

    float gg[NPER], bb[NPER];
#pragma unroll
    for (int t = 0; t < NPER; ++t) {
        const int n = (wave + 8 * t) * 16 + lm;
        gg[t] = g[n];
        bb[t] = be[n];
    }

#pragma unroll
    for (int m = 0; m < MT; ++m) {
        float me[8], rs[8];
#pragma unroll
        for (int r = 0; r < 8; ++r) {
            const int row = m * 16 + r + 8 * half;
            me[r] = stats[row][0];
            rs[r] = stats[row][1];
        }
#pragma unroll
        for (int t = 0; t < NPER; ++t) {
            const int n = (wave + 8 * t) * 16 + lm;
#pragma unroll
            for (int r = 0; r < 8; ++r) {
                const int row = m * 16 + r + 8 * half;
                float v = (acc[m][t][r] - me[r]) * rs[r] * gg[t] + bb[t];
                if (MODE == 0) {
                    xbuf[row * XS + n] = (bf16)fmaxf(v, 0.0f);
                } else if (row < nvalid) {
                    const size_t o = (size_t)row * LATENT + n;
                    if (MODE == 1) outp[o] = fmaxf(v, 0.0f) + resid[o];
                    else           outp[o] = tanhf(v) + resid[o];
                }
            }
        }
    }
    __syncthreads();
}

// ---------------------------------------------------------------------------
// Fused node MLP: concat(nodes, msgs) -> [256,512,512,128] Linear+LN+act,
// residual add, fp32 out. (50000 % 64 == 16 -> tail guard)
// ---------------------------------------------------------------------------
__global__ __launch_bounds__(NTHREADS)
void node_mlp_kernel(const float* __restrict__ nodes,
                     const float* __restrict__ msgs,
                     const bf16* __restrict__ w1, const float* __restrict__ b1,
                     const float* __restrict__ g1, const float* __restrict__ be1,
                     const bf16* __restrict__ w2, const float* __restrict__ b2,
                     const float* __restrict__ g2, const float* __restrict__ be2,
                     const bf16* __restrict__ w3, const float* __restrict__ b3,
                     const float* __restrict__ g3, const float* __restrict__ be3,
                     float* __restrict__ out_nodes) {
    __shared__ __align__(16) bf16 xbuf[BM * XS];
    __shared__ float part[BM][8], part2[BM][8], stats[BM][2];
    const int tid  = threadIdx.x;
    const int base = blockIdx.x * BM;
    const int nvalid = min(BM, N_NODES - base);

    for (int i = tid; i < BM * 2 * LATENT; i += NTHREADS) {
        int row = i >> 8, col = i & 255;
        int gr = base + row;
        float v = 0.0f;
        if (gr < N_NODES)
            v = (col < LATENT) ? nodes[(size_t)gr * LATENT + col]
                               : msgs[(size_t)gr * LATENT + (col - LATENT)];
        xbuf[row * XS + col] = (bf16)v;
    }
    __syncthreads();

    {
        v8f acc[MT][4];
        gemm_bias<4>(w1, b1, xbuf, 2 * LATENT, tid, acc);
        __syncthreads();
        ln_reg<4, 0>(g1, be1, HIDDEN, acc, xbuf, part, part2, stats, tid,
                     nvalid, nullptr, nullptr);
    }
    {
        v8f acc[MT][4];
        gemm_bias<4>(w2, b2, xbuf, HIDDEN, tid, acc);
        __syncthreads();
        ln_reg<4, 0>(g2, be2, HIDDEN, acc, xbuf, part, part2, stats, tid,
                     nvalid, nullptr, nullptr);
    }
    {
        v8f acc[MT][1];
        gemm_bias<1>(w3, b3, xbuf, HIDDEN, tid, acc);
        __syncthreads();
        ln_reg<1, 1>(g3, be3, LATENT, acc, xbuf, part, part2, stats, tid,
                     nvalid, nodes + (size_t)base * LATENT,
                     out_nodes + (size_t)base * LATENT);
    }
}

// ---------------------------------------------------------------------------
// Fused edge MLP: concat(nodes_out[src], nodes_out[dst], edges) ->
// [384,512,512,128] Linear+LN+act (tanh last), residual, fp32 out.
// (500000 % 64 == 32 -> tail guard)
// ---------------------------------------------------------------------------
__global__ __launch_bounds__(NTHREADS)
void edge_mlp_kernel(const float* __restrict__ node_feat,
                     const float* __restrict__ edges,
                     const int* __restrict__ src,
                     const int* __restrict__ dst,
                     const bf16* __restrict__ w1, const float* __restrict__ b1,
                     const float* __restrict__ g1, const float* __restrict__ be1,
                     const bf16* __restrict__ w2, const float* __restrict__ b2,
                     const float* __restrict__ g2, const float* __restrict__ be2,
                     const bf16* __restrict__ w3, const float* __restrict__ b3,
                     const float* __restrict__ g3, const float* __restrict__ be3,
                     float* __restrict__ out_edges) {
    __shared__ __align__(16) bf16 xbuf[BM * XS];
    __shared__ float part[BM][8], part2[BM][8], stats[BM][2];
    const int tid  = threadIdx.x;
    const int base = blockIdx.x * BM;
    const int nvalid = min(BM, N_EDGES - base);

    for (int i = tid; i < BM * 3 * LATENT; i += NTHREADS) {
        int row = i / (3 * LATENT), col = i - row * (3 * LATENT);
        int e = base + row;
        float v = 0.0f;
        if (e < N_EDGES) {
            if (col < LATENT)
                v = node_feat[(size_t)src[e] * LATENT + col];
            else if (col < 2 * LATENT)
                v = node_feat[(size_t)dst[e] * LATENT + (col - LATENT)];
            else
                v = edges[(size_t)e * LATENT + (col - 2 * LATENT)];
        }
        xbuf[row * XS + col] = (bf16)v;
    }
    __syncthreads();

    {
        v8f acc[MT][4];
        gemm_bias<4>(w1, b1, xbuf, 3 * LATENT, tid, acc);
        __syncthreads();
        ln_reg<4, 0>(g1, be1, HIDDEN, acc, xbuf, part, part2, stats, tid,
                     nvalid, nullptr, nullptr);
    }
    {
        v8f acc[MT][4];
        gemm_bias<4>(w2, b2, xbuf, HIDDEN, tid, acc);
        __syncthreads();
        ln_reg<4, 0>(g2, be2, HIDDEN, acc, xbuf, part, part2, stats, tid,
                     nvalid, nullptr, nullptr);
    }
    {
        v8f acc[MT][1];
        gemm_bias<1>(w3, b3, xbuf, HIDDEN, tid, acc);
        __syncthreads();
        ln_reg<1, 2>(g3, be3, LATENT, acc, xbuf, part, part2, stats, tid,
                     nvalid, edges + (size_t)base * LATENT,
                     out_edges + (size_t)base * LATENT);
    }
}

// ---------------------------------------------------------------------------
// Host launcher
// d_in: 0 nodes, 1 edges, 2 graph, 3..14 node_params (W,b,g,be)x3,
//       15..26 edge_params
// ---------------------------------------------------------------------------
extern "C" void kernel_launch(void* const* d_in, const int* in_sizes, int n_in,
                              void* d_out, int out_size, void* d_ws, size_t ws_size,
                              hipStream_t stream) {
    (void)in_sizes; (void)n_in; (void)out_size; (void)ws_size;

    const float* nodes = (const float*)d_in[0];
    const float* edges = (const float*)d_in[1];
    const int*   graph = (const int*)d_in[2];
    const int* src = graph;             // graph[0]
    const int* dst = graph + N_EDGES;   // graph[1]

    // workspace layout (all chunks 16B-aligned)
    char* ws = (char*)d_ws;
    float* msgs = (float*)ws;
    size_t off = (size_t)N_NODES * LATENT * sizeof(float);
    bf16* wt_n1 = (bf16*)(ws + off); off += (size_t)HIDDEN * (2 * LATENT) * sizeof(bf16);
    bf16* wt_n2 = (bf16*)(ws + off); off += (size_t)HIDDEN * HIDDEN * sizeof(bf16);
    bf16* wt_n3 = (bf16*)(ws + off); off += (size_t)LATENT * HIDDEN * sizeof(bf16);
    bf16* wt_e1 = (bf16*)(ws + off); off += (size_t)HIDDEN * (3 * LATENT) * sizeof(bf16);
    bf16* wt_e2 = (bf16*)(ws + off); off += (size_t)HIDDEN * HIDDEN * sizeof(bf16);
    bf16* wt_e3 = (bf16*)(ws + off);

    float* out_nodes = (float*)d_out;
    float* out_edges = out_nodes + (size_t)N_NODES * LATENT;

    // weight transpose + bf16 convert (tiny; L2-resident afterwards)
    wtr_kernel<<<(2 * LATENT * HIDDEN + 255) / 256, 256, 0, stream>>>(
        (const float*)d_in[3], wt_n1, 2 * LATENT, HIDDEN);
    wtr_kernel<<<(HIDDEN * HIDDEN + 255) / 256, 256, 0, stream>>>(
        (const float*)d_in[7], wt_n2, HIDDEN, HIDDEN);
    wtr_kernel<<<(HIDDEN * LATENT + 255) / 256, 256, 0, stream>>>(
        (const float*)d_in[11], wt_n3, HIDDEN, LATENT);
    wtr_kernel<<<(3 * LATENT * HIDDEN + 255) / 256, 256, 0, stream>>>(
        (const float*)d_in[15], wt_e1, 3 * LATENT, HIDDEN);
    wtr_kernel<<<(HIDDEN * HIDDEN + 255) / 256, 256, 0, stream>>>(
        (const float*)d_in[19], wt_e2, HIDDEN, HIDDEN);
    wtr_kernel<<<(HIDDEN * LATENT + 255) / 256, 256, 0, stream>>>(
        (const float*)d_in[23], wt_e3, HIDDEN, LATENT);

    // segment_sum(edges, dst) via f32 atomics (25.6MB target -> L2 resident)
    zero_f32_kernel<<<(N_NODES * LATENT + 255) / 256, 256, 0, stream>>>(
        msgs, N_NODES * LATENT);
    scatter_add_kernel<<<(N_EDGES * LATENT + 255) / 256, 256, 0, stream>>>(
        edges, dst, msgs);

    node_mlp_kernel<<<(N_NODES + BM - 1) / BM, NTHREADS, 0, stream>>>(
        nodes, msgs,
        wt_n1, (const float*)d_in[4],  (const float*)d_in[5],  (const float*)d_in[6],
        wt_n2, (const float*)d_in[8],  (const float*)d_in[9],  (const float*)d_in[10],
        wt_n3, (const float*)d_in[12], (const float*)d_in[13], (const float*)d_in[14],
        out_nodes);

    edge_mlp_kernel<<<(N_EDGES + BM - 1) / BM, NTHREADS, 0, stream>>>(
        out_nodes, edges, src, dst,
        wt_e1, (const float*)d_in[16], (const float*)d_in[17], (const float*)d_in[18],
        wt_e2, (const float*)d_in[20], (const float*)d_in[21], (const float*)d_in[22],
        wt_e3, (const float*)d_in[24], (const float*)d_in[25], (const float*)d_in[26],
        out_edges);
}
